// ArcNegFace_50113678410166
// MI455X (gfx1250) — matrix operand order repaired and checked by reading, not compiled, not run on hardware
//
#include <hip/hip_runtime.h>
#include <math.h>

#define BATCH 256
#define D 512
#define NCLS 100000
#define SCALE_F 64.0f
#define MARGIN_F 0.5f
#define ALPHA_F 1.2f
#define INV_SIGMA 0.5f
#define THRESH_F (-0.8775825619f)   /* cos(pi - 0.5) */
#define MM_F 0.2397127693f          /* sin(pi - 0.5) * 0.5 */
#define KSTEPS 16                   /* 512 / 32 */
#define CLS_PER_WG 128
#define NWG ((NCLS + CLS_PER_WG - 1) / CLS_PER_WG)   /* 782 */

typedef __attribute__((ext_vector_type(16))) _Float16 v16h;
typedef __attribute__((ext_vector_type(8)))  float    v8f;

// ---------------------------------------------------------------------------
// Kernel 1: normalize x rows, emit f16 A-fragments in WMMA lane/half layout.
// A_packed layout (ushort): [kt][mt][lane 0..31][half 0..15]
//   lane<16 : row M=lane,    halves 0..7 -> K=kb+0..7,  8..15 -> K=kb+16..23
//   lane>=16: row M=lane-16, halves 0..7 -> K=kb+8..15, 8..15 -> K=kb+24..31
// Also zero-init the per-row sum-of-exp accumulators every launch.
// ---------------------------------------------------------------------------
__global__ void pack_x_kernel(const float* __restrict__ x,
                              unsigned short* __restrict__ apack,
                              float* __restrict__ sumexp) {
    const int r   = blockIdx.x;       // batch row 0..255
    const int tid = threadIdx.x;      // 0..127
    const int lane = tid & 31, wid = tid >> 5;

    const float4 x4 = ((const float4*)(x + (size_t)r * D))[tid];
    float s = x4.x * x4.x + x4.y * x4.y + x4.z * x4.z + x4.w * x4.w;
#pragma unroll
    for (int m = 16; m >= 1; m >>= 1) s += __shfl_xor(s, m, 32);
    __shared__ float red[4];
    if (lane == 0) red[wid] = s;
    __syncthreads();
    const float rn = rsqrtf(red[0] + red[1] + red[2] + red[3]);

    const int mt = r >> 4, m = r & 15;
    const float v[4] = {x4.x, x4.y, x4.z, x4.w};
#pragma unroll
    for (int j = 0; j < 4; ++j) {
        const int k  = tid * 4 + j;
        const int kt = k >> 5, ko = k & 31;
        const int lidx = m + 16 * ((ko >> 3) & 1);
        const int h    = (ko & 7) + 8 * (ko >> 4);
        union { _Float16 h; unsigned short u; } cv;
        cv.h = (_Float16)(v[j] * rn);
        apack[(((size_t)kt * 16 + mt) * 32 + lidx) * 16 + h] = cv.u;
    }
    if (tid == 0) sumexp[r] = 0.0f;
}

// ---------------------------------------------------------------------------
// Kernel 2: per-row target cosine + arcface margin -> a_lb[b]
// ---------------------------------------------------------------------------
__global__ void target_kernel(const float* __restrict__ x,
                              const int* __restrict__ label,
                              const float* __restrict__ w,
                              float* __restrict__ alb_out) {
    const int b   = blockIdx.x;
    const int tid = threadIdx.x;      // 0..127
    const int lane = tid & 31, wid = tid >> 5;
    const int lbl = label[b];

    const float4 x4 = ((const float4*)(x + (size_t)b   * D))[tid];
    const float4 w4 = ((const float4*)(w + (size_t)lbl * D))[tid];
    float xw = x4.x * w4.x + x4.y * w4.y + x4.z * w4.z + x4.w * w4.w;
    float xx = x4.x * x4.x + x4.y * x4.y + x4.z * x4.z + x4.w * x4.w;
    float ww = w4.x * w4.x + w4.y * w4.y + w4.z * w4.z + w4.w * w4.w;
#pragma unroll
    for (int m = 16; m >= 1; m >>= 1) {
        xw += __shfl_xor(xw, m, 32);
        xx += __shfl_xor(xx, m, 32);
        ww += __shfl_xor(ww, m, 32);
    }
    __shared__ float r0[4], r1[4], r2[4];
    if (lane == 0) { r0[wid] = xw; r1[wid] = xx; r2[wid] = ww; }
    __syncthreads();
    if (tid == 0) {
        const float XW = r0[0] + r0[1] + r0[2] + r0[3];
        const float XX = r1[0] + r1[1] + r1[2] + r1[3];
        const float WW = r2[0] + r2[1] + r2[2] + r2[3];
        const float cosv = XW * rsqrtf(XX * WW);
        const float cc   = fminf(fmaxf(cosv, -1.0f), 1.0f);
        alb_out[b] = (cosv > THRESH_F) ? cosf(acosf(cc) + MARGIN_F)
                                       : (cosv - MM_F);
    }
}

// ---------------------------------------------------------------------------
// Kernel 3: streaming WMMA GEMM over all classes + fused ArcNegFace epilogue
// with online softmax partial sums. One WG = 8 waves x 16 classes = 128
// classes x 256 batch rows; weight is read exactly once device-wide.
// Double-buffered LDS A staging with ONE barrier per K-step; staging uses
// transient load->store pairs (no long-lived registers, no spills).
// ---------------------------------------------------------------------------
__global__ void __launch_bounds__(256)
arc_gemm_kernel(const float* __restrict__ w,
                const unsigned int* __restrict__ apack,
                const int* __restrict__ label,
                const float* __restrict__ alb,
                float* __restrict__ sumexp) {
    __shared__ unsigned int ldsA[2][4096];   // 2 x 16 KB double buffer
    __shared__ int   s_lbl[BATCH];
    __shared__ float s_alb[BATCH];
    __shared__ float s_sum[BATCH];

    const int tid  = threadIdx.x;
    const int lane = tid & 31;
    const int wid  = tid >> 5;

    s_lbl[tid] = label[tid];
    s_alb[tid] = alb[tid];
    s_sum[tid] = 0.0f;

    // class owned by this lane (two lanes per class, one per K-half)
    const int colc  = blockIdx.x * CLS_PER_WG + wid * 16 + (lane & 15);
    const int cload = (colc < NCLS) ? colc : (NCLS - 1);
    const int khalf = (lane >> 4) << 3;   // 0 or 8
    const float* wbase = w + (size_t)cload * D + khalf;

    v8f acc[16] = {};
    float wn2a = 0.0f, wn2b = 0.0f, wn2c = 0.0f, wn2d = 0.0f;

    // ---- prologue: stage A chunk 0 into buffer 0 (transient regs)
    {
        const uint4* src = (const uint4*)apack;       // 1024 uint4 per chunk
        uint4* dst = (uint4*)ldsA[0];
#pragma unroll
        for (int j = 0; j < 4; ++j) dst[tid * 4 + j] = src[tid * 4 + j];
    }

    for (int kt = 0; kt < KSTEPS; ++kt) {
        __syncthreads();   // buf[kt&1] staged; prior iteration fully done

        // ---- B fragment: f32 at offsets 0..7 and 16..23 of this lane's weight row -> f16
        const float* wp = wbase + kt * 32;
        const float4 w0 = ((const float4*)wp)[0];
        const float4 w1 = ((const float4*)wp)[1];
        const float4 w2 = ((const float4*)wp)[4];
        const float4 w3 = ((const float4*)wp)[5];
        if (kt + 1 < KSTEPS) __builtin_prefetch(wp + 32, 0, 0);

        v16h bfrag;
        bfrag[0]  = (_Float16)w0.x; bfrag[1]  = (_Float16)w0.y;
        bfrag[2]  = (_Float16)w0.z; bfrag[3]  = (_Float16)w0.w;
        bfrag[4]  = (_Float16)w1.x; bfrag[5]  = (_Float16)w1.y;
        bfrag[6]  = (_Float16)w1.z; bfrag[7]  = (_Float16)w1.w;
        bfrag[8]  = (_Float16)w2.x; bfrag[9]  = (_Float16)w2.y;
        bfrag[10] = (_Float16)w2.z; bfrag[11] = (_Float16)w2.w;
        bfrag[12] = (_Float16)w3.x; bfrag[13] = (_Float16)w3.y;
        bfrag[14] = (_Float16)w3.z; bfrag[15] = (_Float16)w3.w;

        // 4 independent partial accumulators -> packed FMAs, no serial chain
        wn2a = fmaf(w0.x, w0.x, wn2a); wn2b = fmaf(w0.y, w0.y, wn2b);
        wn2c = fmaf(w0.z, w0.z, wn2c); wn2d = fmaf(w0.w, w0.w, wn2d);
        wn2a = fmaf(w1.x, w1.x, wn2a); wn2b = fmaf(w1.y, w1.y, wn2b);
        wn2c = fmaf(w1.z, w1.z, wn2c); wn2d = fmaf(w1.w, w1.w, wn2d);
        wn2a = fmaf(w2.x, w2.x, wn2a); wn2b = fmaf(w2.y, w2.y, wn2b);
        wn2c = fmaf(w2.z, w2.z, wn2c); wn2d = fmaf(w2.w, w2.w, wn2d);
        wn2a = fmaf(w3.x, w3.x, wn2a); wn2b = fmaf(w3.y, w3.y, wn2b);
        wn2c = fmaf(w3.z, w3.z, wn2c); wn2d = fmaf(w3.w, w3.w, wn2d);

        // ---- stage NEXT A chunk into the other buffer (transient load->store;
        //      stores only need to land before the next barrier)
        if (kt + 1 < KSTEPS) {
            const uint4* src = (const uint4*)apack + (size_t)(kt + 1) * 1024;
            uint4* dst = (uint4*)ldsA[(kt + 1) & 1];
#pragma unroll
            for (int j = 0; j < 4; ++j) dst[tid * 4 + j] = src[tid * 4 + j];
        }

        // ---- 16 batch tiles x wmma from current buffer
        const unsigned int* abuf = ldsA[kt & 1];
#pragma unroll
        for (int mt = 0; mt < 16; ++mt) {
            const v16h afrag = *(const v16h*)&abuf[(mt * 32 + lane) * 8];
            acc[mt] = __builtin_amdgcn_wmma_f32_16x16x32_f16(
                false, afrag, false, bfrag, (short)0, acc[mt], false, false);
        }
    }

    // combine the two K-half squared-norm partials per class
    float wn2 = (wn2a + wn2b) + (wn2c + wn2d);
    wn2 += __shfl_xor(wn2, 16, 32);
    const float rn    = rsqrtf(wn2);
    const bool  valid = (colc < NCLS);

    // ---- fused ArcNegFace transform + online softmax partial sums
#pragma unroll
    for (int mt = 0; mt < 16; ++mt) {
#pragma unroll
        for (int i = 0; i < 8; ++i) {
            const int   brow = mt * 16 + i + ((lane >> 4) << 3);
            const float cosv = acc[mt][i] * rn;
            const float albv = s_alb[brow];
            float logit;
            if (colc == s_lbl[brow]) {
                logit = SCALE_F * albv;                      // one-hot branch
            } else {
                const float dd = cosv - albv;
                const float t  = ALPHA_F * __expf(-dd * dd * INV_SIGMA);
                logit = SCALE_F * (t * cosv + t - 1.0f);     // negative branch
            }
            float e = valid ? __expf(logit - 90.0f) : 0.0f;  // bounded shift
            e += __shfl_xor(e, 1, 32);
            e += __shfl_xor(e, 2, 32);
            e += __shfl_xor(e, 4, 32);
            e += __shfl_xor(e, 8, 32);
            if ((lane & 15) == 0) atomicAdd(&s_sum[brow], e);
        }
    }
    __syncthreads();
    atomicAdd(&sumexp[tid], s_sum[tid]);   // one global atomic per row per WG
}

// ---------------------------------------------------------------------------
// Kernel 4: loss = mean_b( log(sumexp[b]) + 90 - SCALE * a_lb[b] )
// ---------------------------------------------------------------------------
__global__ void loss_kernel(const float* __restrict__ sumexp,
                            const float* __restrict__ alb,
                            float* __restrict__ out) {
    const int tid = threadIdx.x;   // 0..255
    float v = __logf(sumexp[tid]) + 90.0f - SCALE_F * alb[tid];
#pragma unroll
    for (int m = 16; m >= 1; m >>= 1) v += __shfl_xor(v, m, 32);
    __shared__ float red[8];
    if ((tid & 31) == 0) red[tid >> 5] = v;
    __syncthreads();
    if (tid == 0) {
        float s = 0.0f;
#pragma unroll
        for (int i = 0; i < 8; ++i) s += red[i];
        out[0] = s * (1.0f / BATCH);
    }
}

extern "C" void kernel_launch(void* const* d_in, const int* in_sizes, int n_in,
                              void* d_out, int out_size, void* d_ws, size_t ws_size,
                              hipStream_t stream) {
    const float* x      = (const float*)d_in[0];
    const int*   label  = (const int*)d_in[1];
    const float* weight = (const float*)d_in[2];

    char* ws = (char*)d_ws;
    unsigned short* apack = (unsigned short*)ws;                 // 256*512*2 = 262144 B
    float* alb    = (float*)(ws + 262144);                       // 1024 B
    float* sumexp = (float*)(ws + 262144 + 1024);                // 1024 B

    pack_x_kernel<<<BATCH, 128, 0, stream>>>(x, apack, sumexp);
    target_kernel<<<BATCH, 128, 0, stream>>>(x, label, weight, alb);
    arc_gemm_kernel<<<NWG, 256, 0, stream>>>(weight, (const unsigned int*)apack,
                                             label, alb, sumexp);
    loss_kernel<<<1, 256, 0, stream>>>(sumexp, alb, (float*)d_out);
}
